// FISTA_16930761081135
// MI455X (gfx1250) — matrix operand order repaired and checked by reading, not compiled, not run on hardware
//
#include <hip/hip_runtime.h>
#include <cmath>
#include <cstdint>

// ---------------------------------------------------------------------------
// MON/FISTA equilibrium solve on MI455X (gfx1250, wave32, WMMA).
//   Weff = (1-m)I - A^T A + (B^T - B)          (precomputed, 256x256)
//   b    = x @ U^T + ub                         (one 32768x256x256 GEMM)
//   40x:  x+ = relu(0.5*(y + y@Weff + b)); y+ = x+ + beta*(x+ - x)
//   out  = relu(y@Weff + b)
// GEMMs: split-bf16 (hi+lo) v_wmma_f32_16x16x32_bf16, f32 accumulate,
// 4x M-tile register blocking per wave (B-fragments reused, 4 independent
// accumulator chains for XDL pipelining).
// ---------------------------------------------------------------------------

typedef __attribute__((ext_vector_type(16))) __bf16          v16bf;
typedef __attribute__((ext_vector_type(16))) unsigned short  v16u;
typedef __attribute__((ext_vector_type(8)))  unsigned short  v8u;
typedef __attribute__((ext_vector_type(8)))  float           v8f;

constexpr int   BATCH    = 32768;
constexpr int   DD       = 256;
constexpr float MPAR     = 0.2f;
constexpr int   MAX_ITER = 40;
constexpr int   NMT      = 4;    // M-tiles per wave (64 rows)

// ---- bf16 split helpers (RTNE, bit-exact) ----
__device__ __forceinline__ unsigned short f2bf_rtne(float f) {
  union { float f; unsigned u; } c; c.f = f;
  unsigned u = c.u;
  unsigned r = u + 0x7FFFu + ((u >> 16) & 1u);
  return (unsigned short)(r >> 16);
}
__device__ __forceinline__ float bf2f(unsigned short h) {
  union { unsigned u; float f; } c; c.u = ((unsigned)h) << 16;
  return c.f;
}
__device__ __forceinline__ void split2(float f, unsigned short& h, unsigned short& l) {
  h = f2bf_rtne(f);
  l = f2bf_rtne(f - bf2f(h));
}

// ---------------------------------------------------------------------------
__global__ void k_split_bf16(const float* __restrict__ src,
                             unsigned short* __restrict__ hi,
                             unsigned short* __restrict__ lo, int n) {
  int i = blockIdx.x * blockDim.x + threadIdx.x;
  if (i < n) {
    unsigned short h, l;
    split2(src[i], h, l);
    hi[i] = h; lo[i] = l;
  }
}

// ---------------------------------------------------------------------------
// WeffT[n][k] = (1-m)*(n==k) - sum_j A[j][k]*A[j][n] + B[n][k] - B[k][n]
// Stored transposed ([N][K] contiguous) so WMMA B-fragments are 32B loads.
// ---------------------------------------------------------------------------
__global__ void k_weff(const float* __restrict__ A, const float* __restrict__ B,
                       unsigned short* __restrict__ whi, unsigned short* __restrict__ wlo) {
  int n = blockIdx.x;
  int k = threadIdx.x;
  float s = 0.0f;
  for (int j = 0; j < DD; ++j) s += A[j * DD + k] * A[j * DD + n];
  float w = -s + B[n * DD + k] - B[k * DD + n];
  if (n == k) w += (1.0f - MPAR);
  unsigned short h, l;
  split2(w, h, l);
  whi[n * DD + k] = h;
  wlo[n * DD + k] = l;
}

// ---------------------------------------------------------------------------
// 4x 16x16 output tiles per wave: acc[mt] = Z[m0+16mt : +16, :] @ WeffT[n0:+16,:]^T
// Split product: Zh*Wh + Zh*Wl + Zl*Wh  (3 WMMAs / K=32 step / M-tile).
// B-fragments loaded once per K-step, reused across the 4 M-tiles; 4
// independent accumulator chains overlap in the XDL pipe.
// A-frag (16-bit 16x32): lane L -> row L%16; K = 16 contiguous (+16 if L>=16).
// B-frag (16-bit 32x16): lane L -> col L%16; K = 16 contiguous (+16 if L>=16).
// ---------------------------------------------------------------------------
__device__ __forceinline__ void gemm_tile4(const unsigned short* __restrict__ Zh,
                                           const unsigned short* __restrict__ Zl,
                                           const unsigned short* __restrict__ WTh,
                                           const unsigned short* __restrict__ WTl,
                                           int m0, int n0, int lane, v8f acc[NMT]) {
  const int l15 = lane & 15;
  const int lh  = lane >> 4;  // lane half: 0 or 1
  const unsigned short* arh = Zh  + (size_t)(m0 + l15) * DD + lh * 16;
  const unsigned short* arl = Zl  + (size_t)(m0 + l15) * DD + lh * 16;
  const unsigned short* brh = WTh + (size_t)(n0 + l15) * DD + lh * 16;
  const unsigned short* brl = WTl + (size_t)(n0 + l15) * DD + lh * 16;
#pragma unroll
  for (int mt = 0; mt < NMT; ++mt) acc[mt] = v8f{};
#pragma unroll
  for (int ks = 0; ks < DD; ks += 32) {
    v16u bhu = *(const v16u*)(brh + ks);
    v16u blu = *(const v16u*)(brl + ks);
    v16bf BH = __builtin_bit_cast(v16bf, bhu);
    v16bf BL = __builtin_bit_cast(v16bf, blu);
#pragma unroll
    for (int mt = 0; mt < NMT; ++mt) {
      const size_t moff = (size_t)mt * 16 * DD + ks;
      v8u ah0 = *(const v8u*)(arh + moff);
      v8u ah1 = *(const v8u*)(arh + moff + 8);
      v8u al0 = *(const v8u*)(arl + moff);
      v8u al1 = *(const v8u*)(arl + moff + 8);
      v16u ahu = __builtin_shufflevector(ah0, ah1, 0,1,2,3,4,5,6,7, 8,9,10,11,12,13,14,15);
      v16u alu = __builtin_shufflevector(al0, al1, 0,1,2,3,4,5,6,7, 8,9,10,11,12,13,14,15);
      v16bf AH = __builtin_bit_cast(v16bf, ahu);
      v16bf AL = __builtin_bit_cast(v16bf, alu);
      acc[mt] = __builtin_amdgcn_wmma_f32_16x16x32_bf16(false, AH, false, BH, (short)0, acc[mt], false, false);
      acc[mt] = __builtin_amdgcn_wmma_f32_16x16x32_bf16(false, AH, false, BL, (short)0, acc[mt], false, false);
      acc[mt] = __builtin_amdgcn_wmma_f32_16x16x32_bf16(false, AL, false, BH, (short)0, acc[mt], false, false);
    }
  }
}

// C/D layout: lane L, vgpr r  ->  (m_tile_base + r + (L>=16)*8, n0 + L%16)
// Wave -> (64-row M-group, 16-col N-tile):
#define TILE_PROLOGUE()                                              \
  int wave = blockIdx.x * (blockDim.x >> 5) + (threadIdx.x >> 5);    \
  int lane = threadIdx.x & 31;                                       \
  int n0 = (wave & ((DD / 16) - 1)) * 16;                            \
  int m0 = (wave / (DD / 16)) * (16 * NMT);                          \
  int l15 = lane & 15;                                               \
  int lh  = lane >> 4;

// ---------------------------------------------------------------------------
// b = x @ U^T + ub.  U (f32 [n][k]) is already in [N][K] layout for x @ U^T.
// ---------------------------------------------------------------------------
__global__ void k_bias(const unsigned short* __restrict__ Xh,
                       const unsigned short* __restrict__ Xl,
                       const unsigned short* __restrict__ Uh,
                       const unsigned short* __restrict__ Ul,
                       const float* __restrict__ ub,
                       float* __restrict__ bout) {
  TILE_PROLOGUE();
  v8f acc[NMT];
  gemm_tile4(Xh, Xl, Uh, Ul, m0, n0, lane, acc);
  int n = n0 + l15;
  float uv = ub[n];
#pragma unroll
  for (int mt = 0; mt < NMT; ++mt) {
#pragma unroll
    for (int r = 0; r < 8; ++r) {
      int m = m0 + mt * 16 + lh * 8 + r;
      bout[(size_t)m * DD + n] = acc[mt][r] + uv;
    }
  }
}

// ---------------------------------------------------------------------------
// One FISTA iteration (fused GEMM + prox + momentum + resplit):
//   Wy = y @ Weff;  x+ = relu(0.5*(y + Wy + b));  y+ = x+ + beta*(x+ - x)
// y kept as hi/lo bf16 pair (GEMM operand); reconstructed hi+lo elementwise
// (~2^-16 relative, fine vs TOL=1e-4). x updated in place (self-index only).
// ---------------------------------------------------------------------------
__global__ void k_iter(const unsigned short* __restrict__ Yh_in,
                       const unsigned short* __restrict__ Yl_in,
                       unsigned short* __restrict__ Yh_out,
                       unsigned short* __restrict__ Yl_out,
                       float* __restrict__ Xk,
                       const float* __restrict__ Bb,
                       const unsigned short* __restrict__ Wh,
                       const unsigned short* __restrict__ Wl,
                       float beta) {
  TILE_PROLOGUE();
  v8f acc[NMT];
  gemm_tile4(Yh_in, Yl_in, Wh, Wl, m0, n0, lane, acc);
  int n = n0 + l15;
#pragma unroll
  for (int mt = 0; mt < NMT; ++mt) {
#pragma unroll
    for (int r = 0; r < 8; ++r) {
      int m = m0 + mt * 16 + lh * 8 + r;
      size_t idx = (size_t)m * DD + n;
      float y    = bf2f(Yh_in[idx]) + bf2f(Yl_in[idx]);
      float xold = Xk[idx];
      float xnew = fmaxf(0.5f * (y + acc[mt][r] + Bb[idx]), 0.0f);
      float ynew = xnew + beta * (xnew - xold);
      Xk[idx] = xnew;
      unsigned short h, l;
      split2(ynew, h, l);
      Yh_out[idx] = h;
      Yl_out[idx] = l;
    }
  }
}

// ---------------------------------------------------------------------------
// out = relu(y @ Weff + b)
// ---------------------------------------------------------------------------
__global__ void k_final(const unsigned short* __restrict__ Yh,
                        const unsigned short* __restrict__ Yl,
                        const float* __restrict__ Bb,
                        const unsigned short* __restrict__ Wh,
                        const unsigned short* __restrict__ Wl,
                        float* __restrict__ out) {
  TILE_PROLOGUE();
  v8f acc[NMT];
  gemm_tile4(Yh, Yl, Wh, Wl, m0, n0, lane, acc);
  int n = n0 + l15;
#pragma unroll
  for (int mt = 0; mt < NMT; ++mt) {
#pragma unroll
    for (int r = 0; r < 8; ++r) {
      int m = m0 + mt * 16 + lh * 8 + r;
      size_t idx = (size_t)m * DD + n;
      out[idx] = fmaxf(acc[mt][r] + Bb[idx], 0.0f);
    }
  }
}

// ---------------------------------------------------------------------------
extern "C" void kernel_launch(void* const* d_in, const int* in_sizes, int n_in,
                              void* d_out, int out_size, void* d_ws, size_t ws_size,
                              hipStream_t stream) {
  const float* x  = (const float*)d_in[0];
  const float* U  = (const float*)d_in[1];
  const float* ub = (const float*)d_in[2];
  const float* A  = (const float*)d_in[3];
  const float* B  = (const float*)d_in[4];
  float* out = (float*)d_out;

  // ---- workspace carve-up (all slices keep 256B alignment) ----
  char* p = (char*)d_ws;
  const size_t fBD = (size_t)BATCH * DD * sizeof(float);           // 32 MB
  const size_t hBD = (size_t)BATCH * DD * sizeof(unsigned short);  // 16 MB
  const size_t hDD = (size_t)DD * DD * sizeof(unsigned short);     // 128 KB
  float*          bbuf = (float*)p;          p += fBD;
  float*          xk   = (float*)p;          p += fBD;
  unsigned short* xh   = (unsigned short*)p; p += hBD;
  unsigned short* xl   = (unsigned short*)p; p += hBD;
  unsigned short* yh0  = (unsigned short*)p; p += hBD;
  unsigned short* yl0  = (unsigned short*)p; p += hBD;
  unsigned short* yh1  = (unsigned short*)p; p += hBD;
  unsigned short* yl1  = (unsigned short*)p; p += hBD;
  unsigned short* Wh   = (unsigned short*)p; p += hDD;
  unsigned short* Wl   = (unsigned short*)p; p += hDD;
  unsigned short* Uh   = (unsigned short*)p; p += hDD;
  unsigned short* Ul   = (unsigned short*)p; p += hDD;

  const int nBD = BATCH * DD;
  const int nDD = DD * DD;
  dim3 blk(256);  // 8 wave32 waves
  // waves = (M-tiles / NMT) * N-tiles = (2048/4)*16 = 8192 -> 1024 blocks
  dim3 gemmGrid((BATCH / (16 * NMT)) * (DD / 16) / 8);

  // ---- precompute: splits + Weff^T + bias GEMM ----
  k_split_bf16<<<(nBD + 255) / 256, blk, 0, stream>>>(x, xh, xl, nBD);
  k_split_bf16<<<(nDD + 255) / 256, blk, 0, stream>>>(U, Uh, Ul, nDD);
  k_weff<<<DD, DD, 0, stream>>>(A, B, Wh, Wl);
  k_bias<<<gemmGrid, blk, 0, stream>>>(xh, xl, Uh, Ul, ub, bbuf);

  // z0 = 0 for x-iterate and y-iterate (bf16 0x0000 == 0.0f)
  hipMemsetAsync(xk,  0, fBD, stream);
  hipMemsetAsync(yh0, 0, hBD, stream);
  hipMemsetAsync(yl0, 0, hBD, stream);

  // ---- 40 FISTA iterations; t/beta are pure host math (no sync) ----
  double t = 1.0;
  unsigned short *yhi = yh0, *yli = yl0, *yho = yh1, *ylo_ = yl1;
  for (int it = 0; it < MAX_ITER; ++it) {
    double tn = 0.5 * (1.0 + std::sqrt(1.0 + 4.0 * t * t));
    float beta = (float)((t - 1.0) / tn);
    t = tn;
    k_iter<<<gemmGrid, blk, 0, stream>>>(yhi, yli, yho, ylo_, xk, bbuf, Wh, Wl, beta);
    unsigned short* th = yhi; yhi = yho; yho = th;
    unsigned short* tl = yli; yli = ylo_; ylo_ = tl;
  }

  // ---- out = relu(y @ Weff + b) ----
  k_final<<<gemmGrid, blk, 0, stream>>>(yhi, yli, bbuf, Wh, Wl, out);
  (void)in_sizes; (void)n_in; (void)out_size; (void)ws_size;
}
